// Sequence_32134945309299
// MI455X (gfx1250) — matrix operand order repaired and checked by reading, not compiled, not run on hardware
//
#include <hip/hip_runtime.h>

// 2-layer LSTM (H=20), B=8192, T timesteps, via CDNA5 v_wmma_f32_16x16x32_f16.
// Weights = A operand (15 fragments held in VGPRs), h^T = B operand (batch in lanes).
// Gate rows reordered: tiles 0-3 = i,f,g,o units 0-15; tile 4 = [i,f,g,o] x units 16-19.
// K-columns: 0-19 real units, 30 = x slot, 31 = bias slot (bias/input folded into GEMM).
// i/f/o gate rows pre-scaled by 0.5 so sigmoid = fma(0.5, v_tanh, 0.5) (no pre-mul).
// All cross-lane ops are VALU v_permlanex16 (no DS in the time loop).

typedef __attribute__((ext_vector_type(16))) _Float16 v16h;
typedef __attribute__((ext_vector_type(8)))  _Float16 v8h;
typedef __attribute__((ext_vector_type(2)))  __fp16   f16x2;
typedef __attribute__((ext_vector_type(8)))  float    v8f;

#define HREAL 20
#define HP    32          // K padding (one WMMA K-step)
#define GP    80          // 5 gate tiles of 16 rows
#define T_IN  256

// ---- lane<->lane+16 swap, pure VALU ----
__device__ __forceinline__ unsigned xor16u(unsigned v) {
#if __has_builtin(__builtin_amdgcn_permlanex16)
    return (unsigned)__builtin_amdgcn_permlanex16((int)v, (int)v,
                                                  0x76543210, (int)0xfedcba98,
                                                  false, false);
#else
    return (unsigned)__shfl_xor((int)v, 16, 32);
#endif
}
__device__ __forceinline__ float xor16f(float v) {
    return __builtin_bit_cast(float, xor16u(__builtin_bit_cast(unsigned, v)));
}

// ---- activations (native v_tanh_f32) ----
__device__ __forceinline__ float tanh1(float x) {
#if __has_builtin(__builtin_amdgcn_tanhf)
    return __builtin_amdgcn_tanhf(x);
#else
    float e = __expf(2.0f * x);
    return 1.0f - 2.0f * __builtin_amdgcn_rcpf(e + 1.0f);
#endif
}
// sigmoid of the ORIGINAL pregate p, given d = 0.5*p (rows pre-scaled by 0.5):
// sigm(p) = 0.5 + 0.5*tanh(0.5*p) = fma(0.5, tanh(d), 0.5)
__device__ __forceinline__ float sigh(float d) {
#if __has_builtin(__builtin_amdgcn_tanhf)
    return fmaf(0.5f, __builtin_amdgcn_tanhf(d), 0.5f);
#else
    return __builtin_amdgcn_rcpf(1.0f + __expf(-2.0f * d));
#endif
}

__device__ __forceinline__ v8f zero8() {
    v8f v;
#pragma unroll
    for (int i = 0; i < 8; ++i) v[i] = 0.0f;
    return v;
}
__device__ __forceinline__ unsigned pkh(float a, float b) {
    f16x2 p = __builtin_amdgcn_cvt_pkrtz(a, b);
    return __builtin_bit_cast(unsigned, p);
}

// A fragment (16x32 f16): row = reordered gate, K = padded unit.
// lane<16 : K{0..15} of row=lane ; lane>=16 : K{16..31} of row=lane-16.
__device__ __forceinline__ v16h loadA(const _Float16* Wm, int tile, int lane) {
    const _Float16* row = Wm + (tile * 16 + (lane & 15)) * HP;
    const int hi = (lane >> 4) * 16;
    v8h lo = *(const v8h*)(row + hi);
    v8h hh = *(const v8h*)(row + 8 + hi);
    v16h a;
#pragma unroll
    for (int i = 0; i < 8; ++i) { a[i] = lo[i]; a[8 + i] = hh[i]; }
    return a;
}

// B fragment (32x16 f16, col = batch = lane%16, K = unit), built at dword granularity.
// h0[r]: lane<16 -> unit r, lane>=16 -> unit 8+r.  h4[0..3]: units 16..19 (replicated).
// Upper-lane dwords: [h4 pairs, 0, 0 | 0, 0, 0, pk(x, 1)]  (K30 = x, K31 = bias one).
__device__ __forceinline__ v16h buildFrag(const float* h0, const float* h4,
                                          bool upper, float xval) {
    unsigned q[4], s[4];
#pragma unroll
    for (int i = 0; i < 4; ++i) {
        q[i] = pkh(h0[2 * i], h0[2 * i + 1]);
        s[i] = xor16u(q[i]);                       // partner's packed pairs
    }
    const unsigned uh0 = pkh(h4[0], h4[1]);
    const unsigned uh1 = pkh(h4[2], h4[3]);
    const unsigned uxb = pkh(xval, 1.0f);
    union { v16h v; unsigned u[8]; } f;
    f.u[0] = upper ? uh0 : q[0];
    f.u[1] = upper ? uh1 : q[1];
    f.u[2] = upper ? 0u  : q[2];
    f.u[3] = upper ? 0u  : q[3];
    f.u[4] = upper ? 0u  : s[0];
    f.u[5] = upper ? 0u  : s[1];
    f.u[6] = upper ? 0u  : s[2];
    f.u[7] = upper ? uxb : s[3];
    return f.v;
}

// Gate tiles 0-3 (units 0-15): i,f,g,o all land in the same lane/VGPR slot.
// di/df/doo are 0.5x pre-scaled (weight rows), dg is unscaled.
__device__ __forceinline__ void cellA(const v8f& di, const v8f& df, const v8f& dg,
                                      const v8f& doo, float* c, float* hn) {
#pragma unroll
    for (int r = 0; r < 8; ++r) {
        float iv = sigh(di[r]);
        float fv = sigh(df[r]);
        float gv = tanh1(dg[r]);
        float ov = sigh(doo[r]);
        float cn = fv * c[r] + iv * gv;
        c[r] = cn;
        hn[r] = ov * tanh1(cn);
    }
}

// Mixed tile 4: rows = [i16..19, f16..19] (lower lanes) / [g16..19, o16..19] (upper).
// One VALU half-swap; both halves then compute units 16..19 identically.
__device__ __forceinline__ void cellB(const v8f& d4, float* c, float* hn, bool upper) {
    float pr[8];
#pragma unroll
    for (int r = 0; r < 8; ++r) pr[r] = xor16f(d4[r]);
#pragma unroll
    for (int r = 0; r < 4; ++r) {
        float iv = sigh(upper ? pr[r]     : d4[r]);
        float fv = sigh(upper ? pr[4 + r] : d4[4 + r]);
        float gv = tanh1(upper ? d4[r]     : pr[r]);
        float ov = sigh(upper ? d4[4 + r] : pr[4 + r]);
        float cn = fv * c[r] + iv * gv;
        c[r] = cn;
        hn[r] = ov * tanh1(cn);
    }
}

__global__ void __launch_bounds__(128)
lstm_wmma_kernel(const float* __restrict__ input,
                 const float* __restrict__ W_ih1, const float* __restrict__ W_hh1,
                 const float* __restrict__ b_ih1, const float* __restrict__ b_hh1,
                 const float* __restrict__ W_ih2, const float* __restrict__ W_hh2,
                 const float* __restrict__ b_ih2, const float* __restrict__ b_hh2,
                 const float* __restrict__ W_lin, const float* __restrict__ b_lin,
                 float* __restrict__ out, int Tout)
{
    // WA[0]=layer1 (W_hh1 | col30=W_ih1 | col31=b1), WA[1]=W_ih2 (col31=b2), WA[2]=W_hh2
    __shared__ alignas(64) _Float16 WA[3][GP][HP];

    const int tid = threadIdx.x;
    if (tid < GP) {   // one reordered gate row p per thread, all 3 matrices
        const int p = tid;
        int q, u;
        if (p < 64) { q = p >> 4; u = p & 15; }
        else        { const int j = p - 64; q = j >> 2; u = 16 + (j & 3); }
        const int   row = q * HREAL + u;
        const float rs  = (q == 2) ? 1.0f : 0.5f;   // pre-scale i,f,o rows for sigmoid
#pragma unroll 4
        for (int k = 0; k < HP; ++k) {
            float w0 = 0.f, w1 = 0.f, w2 = 0.f;
            if (k < HREAL) {
                w0 = W_hh1[row * HREAL + k];
                w1 = W_ih2[row * HREAL + k];
                w2 = W_hh2[row * HREAL + k];
            } else if (k == 30) {
                w0 = W_ih1[row];                       // input weight (x slot)
            } else if (k == 31) {
                w0 = b_ih1[row] + b_hh1[row];          // bias slot
                w1 = b_ih2[row] + b_hh2[row];
            }
            WA[0][p][k] = (_Float16)(w0 * rs);
            WA[1][p][k] = (_Float16)(w1 * rs);
            WA[2][p][k] = (_Float16)(w2 * rs);
        }
    }
    __syncthreads();

    const int  lane  = tid & 31;
    const int  wave  = tid >> 5;
    const bool upper = lane >= 16;
    const int  b     = (blockIdx.x * 4 + wave) * 16 + (lane & 15);

    // 15 loop-invariant weight fragments -> 120 VGPRs, zero LDS traffic in the loop
    v16h A1[5], A2i[5], A2h[5];
#pragma unroll
    for (int m = 0; m < 5; ++m) {
        A1[m]  = loadA(&WA[0][0][0], m, lane);
        A2i[m] = loadA(&WA[1][0][0], m, lane);
        A2h[m] = loadA(&WA[2][0][0], m, lane);
    }

    // output head: per-lane units 0..15, scalar (SGPR) units 16..19 + bias
    float wl[8];
#pragma unroll
    for (int r = 0; r < 8; ++r) wl[r] = W_lin[(upper ? 8 : 0) + r];
    float wl4[4];
#pragma unroll
    for (int j = 0; j < 4; ++j)
        wl4[j] = __builtin_amdgcn_readfirstlane(W_lin[16 + j]);
    const float blin = __builtin_amdgcn_readfirstlane(b_lin[0]);

    float c1a[8], c2a[8], c1b[4], c2b[4];
#pragma unroll
    for (int r = 0; r < 8; ++r) { c1a[r] = 0.f; c2a[r] = 0.f; }
#pragma unroll
    for (int r = 0; r < 4; ++r) { c1b[r] = 0.f; c2b[r] = 0.f; }

    const float* xrow = input + (size_t)b * T_IN;
    float*       orow = out   + (size_t)b * Tout;

    // initial fragments: h = 0, bias one, x_0 baked into h1f
    float z8[8] = {0.f, 0.f, 0.f, 0.f, 0.f, 0.f, 0.f, 0.f};
    float z4[4] = {0.f, 0.f, 0.f, 0.f};
    v16h h1f = buildFrag(z8, z4, upper, xrow[0]);
    v16h h2f = buildFrag(z8, z4, upper, 0.0f);

    // x pipeline, 2 steps ahead: xA = x_{t+1} (consumed mid-iteration t)
    float xA = xrow[(1 < T_IN) ? 1 : (T_IN - 1)];

    for (int t = 0; t < Tout; ++t) {
        const int tn2 = (t + 2 < T_IN) ? (t + 2) : (T_IN - 1);
        const float xB = xrow[tn2];                 // in flight > 1 full iteration

        float h0n[8], h4n[4];

        // ---- layer 1: 5 WMMAs ----
        v8f d[5];
#pragma unroll
        for (int m = 0; m < 5; ++m)
            d[m] = __builtin_amdgcn_wmma_f32_16x16x32_f16(
                       false, A1[m], false, h1f, (short)0, zero8(), false, false);
        cellA(d[0], d[1], d[2], d[3], c1a, h0n);
        cellB(d[4], c1b, h4n, upper);

        // x_{t+1} baked in now; harmless in layer 2 (its x column is zero there)
        h1f = buildFrag(h0n, h4n, upper, xA);

        // ---- layer 2: 10 WMMAs, C-chained accumulation ----
        v8f e[5];
#pragma unroll
        for (int m = 0; m < 5; ++m) {
            v8f acc = __builtin_amdgcn_wmma_f32_16x16x32_f16(
                          false, A2i[m], false, h1f, (short)0, zero8(), false, false);
            e[m] = __builtin_amdgcn_wmma_f32_16x16x32_f16(
                       false, A2h[m], false, h2f, (short)0, acc, false, false);
        }
        float h0n2[8], h4n2[4];
        cellA(e[0], e[1], e[2], e[3], c2a, h0n2);
        cellB(e[4], c2b, h4n2, upper);

        // ---- output head ----
        float pa = 0.0f;
#pragma unroll
        for (int r = 0; r < 8; ++r) pa = fmaf(h0n2[r], wl[r], pa);
        float p4 = 0.0f;                             // identical in both halves
#pragma unroll
        for (int j = 0; j < 4; ++j) p4 = fmaf(h4n2[j], wl4[j], p4);
        const float tot = pa + xor16f(pa) + p4 + blin;
        if (!upper) orow[t] = tot;

        h2f = buildFrag(h0n2, h4n2, upper, 0.0f);
        xA = xB;
    }
}

extern "C" void kernel_launch(void* const* d_in, const int* in_sizes, int n_in,
                              void* d_out, int out_size, void* d_ws, size_t ws_size,
                              hipStream_t stream) {
    const float* input = (const float*)d_in[0];
    // d_in[1] = future (device int scalar); Tout derived from out_size instead.
    const float* W_ih1 = (const float*)d_in[2];
    const float* W_hh1 = (const float*)d_in[3];
    const float* b_ih1 = (const float*)d_in[4];
    const float* b_hh1 = (const float*)d_in[5];
    const float* W_ih2 = (const float*)d_in[6];
    const float* W_hh2 = (const float*)d_in[7];
    const float* b_ih2 = (const float*)d_in[8];
    const float* b_hh2 = (const float*)d_in[9];
    const float* W_lin = (const float*)d_in[10];
    const float* b_lin = (const float*)d_in[11];

    const int B = 8192;
    const int Tout = out_size / B;     // = 256 + future

    dim3 block(128);                   // 4 wave32 waves, 16 batch rows each
    dim3 grid(B / 64);                 // 128 workgroups
    lstm_wmma_kernel<<<grid, block, 0, stream>>>(
        input, W_ih1, W_hh1, b_ih1, b_hh1,
        W_ih2, W_hh2, b_ih2, b_hh2, W_lin, b_lin,
        (float*)d_out, Tout);
}
